// NT_Xent_7060926234724
// MI455X (gfx1250) — matrix operand order repaired and checked by reading, not compiled, not run on hardware
//
#include <hip/hip_runtime.h>
#include <stdint.h>

#define B_ 2048
#define N_ 4096
#define D_ 256
#define TOPK_ 64
#define INV_TEMP 2.0f          // 1 / 0.5
#define EPS_ 1e-8f
#define NEG_INF_ (-3.4e38f)

typedef __attribute__((ext_vector_type(16))) __bf16 v16bf;
typedef __attribute__((ext_vector_type(8)))  float  v8f;

union Frag16 {
    v16bf    v;
    uint32_t u[8];
};

__device__ __forceinline__ uint16_t f2bf(float f) {
    uint32_t u = __float_as_uint(f);
    uint32_t r = u + 0x7FFFu + ((u >> 16) & 1u);   // round-to-nearest-even
    return (uint16_t)(r >> 16);
}
__device__ __forceinline__ float bf2f(uint16_t h) {
    return __uint_as_float(((uint32_t)h) << 16);
}

// ---------------------------------------------------------------------------
// K1: row-normalize z = [z_i; z_j], split each normalized value into bf16
//     hi + lo so that hi + lo ~= fp32 value.  One block per row.
// ---------------------------------------------------------------------------
__global__ void __launch_bounds__(256)
k_normalize(const float* __restrict__ zi, const float* __restrict__ zj,
            uint16_t* __restrict__ znh, uint16_t* __restrict__ znl) {
    const int row = blockIdx.x;
    const int t   = threadIdx.x;
    const float* src = (row < B_) ? (zi + (size_t)row * D_)
                                  : (zj + (size_t)(row - B_) * D_);
    float v = src[t];
    __shared__ float red[256];
    red[t] = v * v;
    __syncthreads();
    for (int s = 128; s > 0; s >>= 1) {
        if (t < s) red[t] += red[t + s];
        __syncthreads();
    }
    float inv = 1.0f / fmaxf(sqrtf(red[0]), EPS_);
    float zn  = v * inv;
    uint16_t h = f2bf(zn);
    float    lo = zn - bf2f(h);
    znh[(size_t)row * D_ + t] = h;
    znl[(size_t)row * D_ + t] = f2bf(lo);
}

// ---------------------------------------------------------------------------
// K2: sim = (zn @ zn^T) * INV_TEMP via bf16 WMMA with hi/lo error split.
//     Block = 256 threads = 8 waves; one 16-row A strip in LDS, one 16x16
//     tile per wave; 3 chained v_wmma_f32_16x16x32_bf16 per 32-K step.
// ---------------------------------------------------------------------------
__global__ void __launch_bounds__(256)
k_simgemm(const uint16_t* __restrict__ znh, const uint16_t* __restrict__ znl,
          float* __restrict__ sim) {
    __shared__ uint16_t Ah[16 * D_];   // 8 KB
    __shared__ uint16_t Al[16 * D_];   // 8 KB

    const int tid    = threadIdx.x;
    const int lane   = tid & 31;
    const int wave   = tid >> 5;
    const int m_tile = blockIdx.x >> 5;                 // 0..255
    const int n_tile = ((blockIdx.x & 31) << 3) + wave; // 0..255
    const int m0     = m_tile << 4;

    {   // cooperative staging of the A strip (16 rows x 256 K), hi + lo
        const int row = tid >> 4, seg = tid & 15;       // seg*16 bf16 = 32B
        const uint4* sh = (const uint4*)(znh + (size_t)(m0 + row) * D_ + seg * 16);
        const uint4* sl = (const uint4*)(znl + (size_t)(m0 + row) * D_ + seg * 16);
        uint4* dh = (uint4*)(&Ah[row * D_ + seg * 16]);
        uint4* dl = (uint4*)(&Al[row * D_ + seg * 16]);
        dh[0] = sh[0]; dh[1] = sh[1];
        dl[0] = sl[0]; dl[1] = sl[1];
    }
    __syncthreads();

    const int    half     = lane >> 4;
    const int    mrow     = lane & 15;
    const int    ncol     = (n_tile << 4) + (lane & 15);
    const size_t browbase = (size_t)ncol * D_ + half * 16;   // B: lanes 0-15 K0-15, 16-31 K16-31

    v8f acc = {};
#pragma unroll
    for (int ks = 0; ks < D_ / 32; ++ks) {
        const int k0 = ks * 32;

        // A fragment (16x32 bf16): lanes 0-15 K0-15, 16-31 K16-31 (matches B)
        Frag16 ah, al;
#pragma unroll
        for (int v = 0; v < 8; ++v) {
            const int kk = 2 * v + 16 * half;
            ah.u[v] = *(const uint32_t*)&Ah[mrow * D_ + k0 + kk];
            al.u[v] = *(const uint32_t*)&Al[mrow * D_ + k0 + kk];
        }

        // B fragment (32x16 bf16): 2 x b128 contiguous per lane from L2
        const uint4 h0 = ((const uint4*)(znh + browbase + k0))[0];
        const uint4 h1 = ((const uint4*)(znh + browbase + k0))[1];
        const uint4 l0 = ((const uint4*)(znl + browbase + k0))[0];
        const uint4 l1 = ((const uint4*)(znl + browbase + k0))[1];
        Frag16 bh, bl;
        bh.u[0] = h0.x; bh.u[1] = h0.y; bh.u[2] = h0.z; bh.u[3] = h0.w;
        bh.u[4] = h1.x; bh.u[5] = h1.y; bh.u[6] = h1.z; bh.u[7] = h1.w;
        bl.u[0] = l0.x; bl.u[1] = l0.y; bl.u[2] = l0.z; bl.u[3] = l0.w;
        bl.u[4] = l1.x; bl.u[5] = l1.y; bl.u[6] = l1.z; bl.u[7] = l1.w;

        if (ks + 1 < D_ / 32) {
            __builtin_prefetch(znh + browbase + k0 + 32, 0, 1);
            __builtin_prefetch(znl + browbase + k0 + 32, 0, 1);
        }

        // hi*hi + hi*lo + lo*hi  (drop lo*lo, ~2^-16 scale)
        acc = __builtin_amdgcn_wmma_f32_16x16x32_bf16(false, ah.v, false, bh.v,
                                                      (short)0, acc, false, false);
        acc = __builtin_amdgcn_wmma_f32_16x16x32_bf16(false, ah.v, false, bl.v,
                                                      (short)0, acc, false, false);
        acc = __builtin_amdgcn_wmma_f32_16x16x32_bf16(false, al.v, false, bh.v,
                                                      (short)0, acc, false, false);
    }

    // C/D layout: VGPR r -> M = r + 8*half, N = lane&15
#pragma unroll
    for (int r = 0; r < 8; ++r) {
        const int m = m0 + r + 8 * half;
        const int n = (n_tile << 4) + (lane & 15);
        sim[(size_t)m * N_ + n] = acc[r] * INV_TEMP;
    }
}

// ---------------------------------------------------------------------------
// K3a: per-row top-64 candidate indices in exact descending (value, then
//      smallest index) order -- the same order as stable argsort(-sim).
//      Self (i) and positive partner (i^B) are excluded (never selectable).
//      Row lives in registers (16 values/lane) with a removal bitmask, so
//      64 successive argmax passes are pure VALU + LDS tree reductions.
// ---------------------------------------------------------------------------
__global__ void __launch_bounds__(256)
k_topk(const float* __restrict__ sim, int* __restrict__ topi) {
    __shared__ float rv[256];
    __shared__ int   rj[256];
    const int i = blockIdx.x, t = threadIdx.x;
    const int ip = i ^ B_;
    const float* srow = sim + (size_t)i * N_;

    float lv[16];
#pragma unroll
    for (int q = 0; q < 16; ++q) lv[q] = srow[(q << 8) + t];

    uint32_t removed = 0u;
    if ((i  & 255) == t) removed |= 1u << (i  >> 8);
    if ((ip & 255) == t) removed |= 1u << (ip >> 8);

    for (int k = 0; k < TOPK_; ++k) {
        float bv = NEG_INF_; int bj = -1;
#pragma unroll
        for (int q = 0; q < 16; ++q) {          // ascending q = ascending j
            const float v = lv[q];
            if (!((removed >> q) & 1u) && v > bv) { bv = v; bj = (q << 8) + t; }
        }
        rv[t] = bv; rj[t] = bj;
        __syncthreads();
        for (int s = 128; s > 0; s >>= 1) {
            if (t < s) {
                const float v2 = rv[t + s]; const int j2 = rj[t + s];
                if (v2 > rv[t] ||
                    (v2 == rv[t] && j2 >= 0 && (rj[t] < 0 || j2 < rj[t]))) {
                    rv[t] = v2; rj[t] = j2;
                }
            }
            __syncthreads();
        }
        const int w = rj[0];                    // winner (always >= 0)
        if ((w & 255) == t) removed |= 1u << (w >> 8);
        if (t == 0) topi[(size_t)i * TOPK_ + k] = w;
        __syncthreads();
    }
}

// ---------------------------------------------------------------------------
// K3b: sequential greedy pairing (faithful to reference).  Single block.
//      For each unpaired i, the chosen j is the first *valid* candidate in
//      the row's exact descending order.  If any of the top-64 is valid,
//      that first-valid candidate IS the global masked argmax (everything
//      outside the list is majorized by the list, stable tie order
//      included).  Only when all 64 are already paired (rare, end-game)
//      do we fall back to the exact full-row scan.
// ---------------------------------------------------------------------------
__global__ void __launch_bounds__(256)
k_greedy(const float* __restrict__ sim, const int* __restrict__ topi,
         int* __restrict__ pa, int* __restrict__ pb) {
    __shared__ unsigned char inp[N_];
    __shared__ short pas[N_], pbs[N_];
    __shared__ float rv[256];
    __shared__ int   rj[256];
    __shared__ int   sel;
    const int t = threadIdx.x;

    for (int j = t; j < N_; j += 256) { inp[j] = 0; pas[j] = (short)j; pbs[j] = (short)j; }
    __syncthreads();

    for (int i = 0; i < N_; ++i) {
        const int ip = i ^ B_;
        if ((inp[i] | inp[ip]) != 0) continue;       // uniform: no syncs skipped

        // phase 1: probe precomputed candidates (exact descending order)
        if (t < TOPK_) {
            const int cj = topi[(size_t)i * TOPK_ + t];
            rj[t] = inp[cj] ? -1 : cj;               // inp[cj]==inp[cj^B] invariant
        }
        __syncthreads();
        if (t == 0) {
            int s = -1;
            for (int q = 0; q < TOPK_; ++q) {
                const int c = rj[q];
                if (c >= 0) { s = c; break; }
            }
            sel = s;
        }
        __syncthreads();
        int j = sel;

        if (j < 0) {
            // phase 2 (rare): exact full-row masked argmax fallback
            float bv = NEG_INF_; int bj = -1;
            const float* srow = sim + (size_t)i * N_;
            for (int jj = t; jj < N_; jj += 256) {
                if (jj == i || jj == ip) continue;
                if (inp[jj]) continue;
                const float v = srow[jj];
                if (v > bv) { bv = v; bj = jj; }     // '>' keeps smallest j on tie
            }
            rv[t] = bv; rj[t] = bj;
            __syncthreads();
            for (int s = 128; s > 0; s >>= 1) {
                if (t < s) {
                    const float v2 = rv[t + s]; const int j2 = rj[t + s];
                    if (v2 > rv[t] ||
                        (v2 == rv[t] && j2 >= 0 && (rj[t] < 0 || j2 < rj[t]))) {
                        rv[t] = v2; rj[t] = j2;
                    }
                }
                __syncthreads();
            }
            j = rj[0];
        }

        if (t == 0 && j >= 0) {
            const int jp = j ^ B_;
            inp[i] = 1; inp[ip] = 1; inp[j] = 1; inp[jp] = 1;
            pas[i] = pas[ip] = (short)j;  pbs[i] = pbs[ip] = (short)jp;
            pas[j] = pas[jp] = (short)i;  pbs[j] = pbs[jp] = (short)ip;
        }
        __syncthreads();
    }
    for (int j = t; j < N_; j += 256) { pa[j] = pas[j]; pb[j] = pbs[j]; }
}

// ---------------------------------------------------------------------------
// K4: per-row loss = logsumexp([pos, first N-4 kept negatives]) - pos.
//     Kept column j: j not in E = dedup{i, i^B, pa[i], pb[i]} and
//     rank(j) = j - |{e in E : e < j}| < N-4  (matches stable-argsort take).
// ---------------------------------------------------------------------------
__global__ void __launch_bounds__(256)
k_rowloss(const float* __restrict__ sim, const int* __restrict__ pa,
          const int* __restrict__ pb, float* __restrict__ rowloss) {
    __shared__ float row[N_];     // 16 KB
    __shared__ float red[256];
    const int i = blockIdx.x, t = threadIdx.x;
    const float* srow = sim + (size_t)i * N_;
    for (int j = t; j < N_; j += 256) row[j] = srow[j];
    __syncthreads();

    const int ip = i ^ B_;
    int E[4]; int m = 0;
    { const int cand[4] = { i, ip, pa[i], pb[i] };
      for (int c = 0; c < 4; ++c) {
          bool dup = false;
          for (int q = 0; q < m; ++q) dup = dup || (E[q] == cand[c]);
          if (!dup) E[m++] = cand[c];
      } }
    const float pos = row[ip];

    float lmax = (t == 0) ? pos : NEG_INF_;
    for (int j = t; j < N_; j += 256) {
        bool kept = true; int less = 0;
        for (int q = 0; q < m; ++q) { if (E[q] == j) kept = false; if (E[q] < j) ++less; }
        if (kept && (j - less) < (N_ - 4)) lmax = fmaxf(lmax, row[j]);
    }
    red[t] = lmax; __syncthreads();
    for (int s = 128; s > 0; s >>= 1) { if (t < s) red[t] = fmaxf(red[t], red[t + s]); __syncthreads(); }
    const float M = red[0];
    __syncthreads();

    float lsum = (t == 0) ? expf(pos - M) : 0.0f;
    for (int j = t; j < N_; j += 256) {
        bool kept = true; int less = 0;
        for (int q = 0; q < m; ++q) { if (E[q] == j) kept = false; if (E[q] < j) ++less; }
        if (kept && (j - less) < (N_ - 4)) lsum += expf(row[j] - M);
    }
    red[t] = lsum; __syncthreads();
    for (int s = 128; s > 0; s >>= 1) { if (t < s) red[t] += red[t + s]; __syncthreads(); }
    if (t == 0) rowloss[i] = (M + logf(red[0])) - pos;
}

// ---------------------------------------------------------------------------
// K5: loss = sum(rowloss) / N
// ---------------------------------------------------------------------------
__global__ void __launch_bounds__(256)
k_finalize(const float* __restrict__ rowloss, float* __restrict__ out) {
    __shared__ float red[256];
    const int t = threadIdx.x;
    float s = 0.0f;
    for (int j = t; j < N_; j += 256) s += rowloss[j];
    red[t] = s; __syncthreads();
    for (int st = 128; st > 0; st >>= 1) { if (t < st) red[t] += red[t + st]; __syncthreads(); }
    if (t == 0) out[0] = red[0] / (float)N_;
}

// ---------------------------------------------------------------------------
extern "C" void kernel_launch(void* const* d_in, const int* in_sizes, int n_in,
                              void* d_out, int out_size, void* d_ws, size_t ws_size,
                              hipStream_t stream) {
    (void)in_sizes; (void)n_in; (void)out_size; (void)ws_size;
    const float* zi = (const float*)d_in[0];
    const float* zj = (const float*)d_in[1];

    // workspace layout (all 16B-aligned):
    //   sim      : N*N f32        = 64 MB
    //   znh, znl : N*D bf16       =  2 MB each
    //   topi     : N*64 int       =  1 MB
    //   pa, pb   : N int
    //   rowloss  : N f32
    char* ws = (char*)d_ws;
    float*    sim = (float*)ws;
    uint16_t* znh = (uint16_t*)(ws + (size_t)N_ * N_ * sizeof(float));
    uint16_t* znl = znh + (size_t)N_ * D_;
    int*      topi = (int*)(znl + (size_t)N_ * D_);
    int*      pa  = topi + (size_t)N_ * TOPK_;
    int*      pb  = pa + N_;
    float*    rowloss = (float*)(pb + N_);
    float*    out = (float*)d_out;

    k_normalize<<<N_, 256, 0, stream>>>(zi, zj, znh, znl);
    k_simgemm<<<(N_ / 16) * (N_ / 16 / 8), 256, 0, stream>>>(znh, znl, sim);
    k_topk<<<N_, 256, 0, stream>>>(sim, topi);
    k_greedy<<<1, 256, 0, stream>>>(sim, topi, pa, pb);
    k_rowloss<<<N_, 256, 0, stream>>>(sim, pa, pb, rowloss);
    k_finalize<<<1, 256, 0, stream>>>(rowloss, out);
}
